// GraphRecommendationModel_33363305956129
// MI455X (gfx1250) — matrix operand order repaired and checked by reading, not compiled, not run on hardware
//
#include <hip/hip_runtime.h>
#include <hip/hip_bf16.h>
#include <math.h>

// ---------------- types ----------------
typedef __attribute__((ext_vector_type(16))) __bf16 bf16x16;
typedef __attribute__((ext_vector_type(8)))  float  f32x8;
typedef unsigned int u32x4 __attribute__((ext_vector_type(4)));
typedef int          i32x4 __attribute__((ext_vector_type(4)));
typedef int          i32x8 __attribute__((ext_vector_type(8)));

union Frag { u32x4 q[2]; bf16x16 v; };

// native bf16 converts (lower to HW cvt on gfx1250; RNE semantics)
__device__ __forceinline__ unsigned short f2bf(float f) {
    union { __bf16 h; unsigned short u; } c;
    c.h = (__bf16)f;
    return c.u;
}
__device__ __forceinline__ unsigned cvt2bf(float a, float b) {
    union { __bf16 h[2]; unsigned u; } c;
    c.h[0] = (__bf16)a;
    c.h[1] = (__bf16)b;
    return c.u;
}
__device__ __forceinline__ float bf2f(unsigned short h) {
    union { unsigned u; float f; } v; v.u = ((unsigned)h) << 16; return v.f;
}

#define B_    256
#define L_    50
#define K_    100
#define H_    64
#define MROWS 112            // K_=100 padded to 7*16
#define MT    7

// packed-weight sizes (bytes)
#define PWA_B 24576          // 4*6*512 bf16
#define PW1_B 9216           // 3*3*512 bf16
#define PW2_B 2048           // 2*1*512 bf16
#define PW_TOTAL (PWA_B + PW1_B + PW2_B)   // 35840 B = 8960 dwords

// LDS layout (dynamic)
#define SA_OFF  0
#define SH1_OFF 28672
#define SH2_OFF (28672 + 21504)            // 50176
#define SWB_OFF (28672 + 21504 + 14336)    // 64512
#define SMEM_TOTAL (SWB_OFF + PW_TOTAL)    // 100352 B

// ---------------- weight pre-pack (B-fragment order, bf16) ----------------
// out[((kc*NT + nt)*32 + lane)*16 + t]
// B 32x16 bf16 fragment: lane half h: elems 0..7 at K=8h+(0..7), elems 8..15 at K=16+8h+(0..7).
__global__ void pack_b_kernel(const float* __restrict__ W, unsigned short* __restrict__ out,
                              int Ksrc, int N, int NT, int total) {
    int i = blockIdx.x * blockDim.x + threadIdx.x;
    if (i >= total) return;
    int t    = i & 15;
    int lane = (i >> 4) & 31;
    int tile = i >> 9;
    int kc   = tile / NT;
    int nt   = tile % NT;
    int khalf = lane >> 4;
    int k = kc * 32 + khalf * 8 + (t & 7) + ((t >> 3) << 4);
    int n = nt * 16 + (lane & 15);
    float v = (k < Ksrc && n < N) ? W[k * N + n] : 0.f;
    out[i] = f2bf(v);
}

// ---------------- fragment loaders (both from LDS) ----------------
// A 16x32 bf16: lane l: row = m0+(l&15); elems 0..7 at K = kc*32 + (l>>4)*8 + (0..7),
//                              elems 8..15 at K = kc*32 + 16 + (l>>4)*8 + (0..7)
__device__ __forceinline__ Frag loadA(const unsigned short* S, int stride, int m0, int kc) {
    int lane  = threadIdx.x & 31;
    int base  = (m0 + (lane & 15)) * stride + kc * 32 + ((lane >> 4) << 3);
    Frag f;
    f.q[0] = *reinterpret_cast<const u32x4*>(S + base);
    f.q[1] = *reinterpret_cast<const u32x4*>(S + base + 16);
    return f;
}
__device__ __forceinline__ Frag loadB(const unsigned short* S, int tileIdx) {
    int lane = threadIdx.x & 31;
    const unsigned short* p = S + ((size_t)(tileIdx * 32 + lane)) * 16;
    Frag f;
    f.q[0] = *reinterpret_cast<const u32x4*>(p);
    f.q[1] = *reinterpret_cast<const u32x4*>(p + 8);
    return f;
}

// ---------------- main fused kernel: one workgroup per (b,l) ----------------
__global__ __launch_bounds__(256)
void edge_mlp_kernel(const int* __restrict__ history,
                     const float* __restrict__ item_table,
                     const float* __restrict__ neighbor,
                     const unsigned int* __restrict__ pWall,   // packed Wa|W1|W2, contiguous
                     const float* __restrict__ ba,
                     const float* __restrict__ b1,
                     const float* __restrict__ b2,
                     const float* __restrict__ W3,
                     const float* __restrict__ b3,
                     float* __restrict__ aggBuf) {
    extern __shared__ char smem[];
    unsigned short* sA  = (unsigned short*)(smem + SA_OFF);   // [112][128] bf16
    unsigned short* sH1 = (unsigned short*)(smem + SH1_OFF);  // [112][96]  bf16
    unsigned short* sH2 = (unsigned short*)(smem + SH2_OFF);  // [112][64]  bf16
    unsigned short* sWb = (unsigned short*)(smem + SWB_OFF);  // packed weights (bf16)
    float* sH3 = (float*)sH1;                                 // [112][16] f32, aliases dead sH1

    const unsigned short* sWa = sWb;
    const unsigned short* sW1 = sWb + PWA_B / 2;
    const unsigned short* sW2 = sWb + (PWA_B + PW1_B) / 2;

    __shared__ unsigned short sHist[H_];
    __shared__ float sLog[MROWS];
    __shared__ float sWgt[MROWS];
    __shared__ float sRed[128];
    __shared__ float sAggP[4 * H_];

    const int tid  = threadIdx.x;
    const int wave = tid >> 5;
    const int lane = tid & 31;
    const int bl   = blockIdx.x;                 // 0 .. B*L-1

    // ---- Stage packed weights global -> LDS (TDM if available) ----
#if __has_builtin(__builtin_amdgcn_tensor_load_to_lds)
    if (wave == 0) {
        unsigned ldsOff = (unsigned)(size_t)(const void*)sWb;     // LDS byte offset (addr[31:0])
        unsigned long long ga = (unsigned long long)(size_t)pWall;
        u32x4 g0;
        g0[0] = 1u;                                               // count=1 (valid descriptor)
        g0[1] = ldsOff;                                           // lds_addr
        g0[2] = (unsigned)(ga & 0xFFFFFFFFu);                     // global_addr[31:0]
        g0[3] = (unsigned)((ga >> 32) & 0x1FFFFFFu) | (2u << 30); // global_addr[56:32] | type=2
        i32x8 g1;
        g1[0] = 0x20000;                 // data_size=2 (4B), wg_mask=0
        g1[1] = (int)(8960u << 16);      // tensor_dim0[15:0]  @bits[63:48]
        g1[2] = (int)(1u << 16);         // tensor_dim0 hi=0; tensor_dim1=1 @bits[95:80]
        g1[3] = (int)(8960u << 16);      // tile_dim0=8960     @bits[127:112]
        g1[4] = 1;                       // tile_dim1=1, tile_dim2=0
        g1[5] = 8960;                    // tensor_dim0_stride[31:0]
        g1[6] = 0;
        g1[7] = 0;
        i32x4 z4 = {};
        i32x8 z8 = {};
        __builtin_amdgcn_tensor_load_to_lds(g0, g1, z4, z4, z8, 0);   // 6-arg form
        __builtin_amdgcn_s_wait_tensorcnt(0);
    }
#else
    {   // cooperative fallback copy (8960 dwords)
        u32x4* d = (u32x4*)sWb;
        const u32x4* s = (const u32x4*)pWall;
        for (int i = tid; i < PW_TOTAL / 16; i += 256) d[i] = s[i];
    }
#endif

    // ---- Phase A: build attn_in [112][128] bf16 in LDS ----
    {
        const float* hrow = item_table + (size_t)history[bl] * H_;
        if (tid < H_) sHist[tid] = f2bf(hrow[tid]);

        const float4* nbv = reinterpret_cast<const float4*>(neighbor + (size_t)bl * K_ * H_);
        for (int e = tid; e < K_ * 16; e += 256) {           // 16 float4 per neighbor row
            int row = e >> 4, c4 = (e & 15) << 2;
            float4 v = nbv[e];
            unsigned* d = (unsigned*)(sA + row * 128 + 64 + c4);
            d[0] = cvt2bf(v.x, v.y);
            d[1] = cvt2bf(v.z, v.w);
        }
        // zero-pad sH2 columns 48..63 (K padded to 64 for layer 3) while loads are in flight
        for (int e = tid; e < MROWS * 16; e += 256)
            sH2[(e >> 4) * 64 + 48 + (e & 15)] = 0;
        __syncthreads();
        for (int e = tid; e < MROWS * 16; e += 256) {
            int row = e >> 4, c4 = (e & 15) << 2;
            unsigned short* d = sA + row * 128 + c4;
            if (row < K_) {
                d[0] = sHist[c4]; d[1] = sHist[c4 + 1]; d[2] = sHist[c4 + 2]; d[3] = sHist[c4 + 3];
            } else {
                d[0] = d[1] = d[2] = d[3] = 0;
                unsigned short* z = sA + row * 128 + 64 + c4;
                z[0] = z[1] = z[2] = z[3] = 0;
            }
        }
        __syncthreads();
    }

    // ---- Layer 1: [112x128] @ [128x96], relu ----  (42 tiles, 4 k-chunks)
    for (int tile = wave; tile < MT * 6; tile += 8) {
        int mt = tile / 6, nt = tile % 6;
        int m0 = mt * 16, n0 = nt * 16;
        f32x8 acc = {};
        #pragma unroll
        for (int kc = 0; kc < 4; ++kc) {
            Frag a = loadA(sA, 128, m0, kc);
            Frag b = loadB(sWa, kc * 6 + nt);
            acc = __builtin_amdgcn_wmma_f32_16x16x32_bf16(false, a.v, false, b.v,
                                                          (short)0, acc, false, false);
        }
        int ncol = n0 + (lane & 15);
        int mb   = m0 + ((lane >> 4) << 3);
        float bias = ba[ncol];
        #pragma unroll
        for (int r = 0; r < 8; r += 2) {
            unsigned u = cvt2bf(fmaxf(acc[r] + bias, 0.f), fmaxf(acc[r + 1] + bias, 0.f));
            sH1[(mb + r) * 96 + ncol]     = (unsigned short)(u & 0xFFFFu);
            sH1[(mb + r + 1) * 96 + ncol] = (unsigned short)(u >> 16);
        }
    }
    __syncthreads();

    // ---- Layer 2: [112x96] @ [96x48], relu ----  (21 tiles, 3 k-chunks)
    for (int tile = wave; tile < MT * 3; tile += 8) {
        int mt = tile / 3, nt = tile % 3;
        int m0 = mt * 16, n0 = nt * 16;
        f32x8 acc = {};
        #pragma unroll
        for (int kc = 0; kc < 3; ++kc) {
            Frag a = loadA(sH1, 96, m0, kc);
            Frag b = loadB(sW1, kc * 3 + nt);
            acc = __builtin_amdgcn_wmma_f32_16x16x32_bf16(false, a.v, false, b.v,
                                                          (short)0, acc, false, false);
        }
        int ncol = n0 + (lane & 15);
        int mb   = m0 + ((lane >> 4) << 3);
        float bias = b1[ncol];
        #pragma unroll
        for (int r = 0; r < 8; r += 2) {
            unsigned u = cvt2bf(fmaxf(acc[r] + bias, 0.f), fmaxf(acc[r + 1] + bias, 0.f));
            sH2[(mb + r) * 64 + ncol]     = (unsigned short)(u & 0xFFFFu);
            sH2[(mb + r + 1) * 64 + ncol] = (unsigned short)(u >> 16);
        }
    }
    __syncthreads();

    // ---- Layer 3: [112x64] @ [64x16], relu, fold W3 ----  (7 tiles, 2 k-chunks)
    for (int mt = wave; mt < MT; mt += 8) {
        int m0 = mt * 16;
        f32x8 acc = {};
        #pragma unroll
        for (int kc = 0; kc < 2; ++kc) {
            Frag a = loadA(sH2, 64, m0, kc);
            Frag b = loadB(sW2, kc);
            acc = __builtin_amdgcn_wmma_f32_16x16x32_bf16(false, a.v, false, b.v,
                                                          (short)0, acc, false, false);
        }
        int n  = lane & 15;
        int mb = m0 + ((lane >> 4) << 3);
        float bias = b2[n];
        float w3v  = W3[n];
        #pragma unroll
        for (int r = 0; r < 8; ++r)
            sH3[(mb + r) * 16 + n] = fmaxf(acc[r] + bias, 0.f) * w3v;
    }
    __syncthreads();

    // ---- logits (sum the 16 pre-scaled columns) ----
    if (tid < MROWS) {
        float s = b3[0];
        #pragma unroll
        for (int n = 0; n < 16; ++n) s += sH3[tid * 16 + n];
        sLog[tid] = s;
    }
    __syncthreads();

    // ---- softmax over 100 neighbors ----
    float lv = (tid < K_) ? sLog[tid] : -3.0e30f;
    if (tid < 128) sRed[tid] = lv;
    __syncthreads();
    for (int s = 64; s > 0; s >>= 1) {
        if (tid < s) sRed[tid] = fmaxf(sRed[tid], sRed[tid + s]);
        __syncthreads();
    }
    float mx = sRed[0];
    __syncthreads();
    float ev = (tid < K_) ? __expf(sLog[tid] - mx) : 0.f;
    if (tid < 128) sRed[tid] = ev;
    __syncthreads();
    for (int s = 64; s > 0; s >>= 1) {
        if (tid < s) sRed[tid] += sRed[tid + s];
        __syncthreads();
    }
    float inv = 1.f / sRed[0];
    if (tid < K_) sWgt[tid] = ev * inv;
    __syncthreads();

    // ---- weighted aggregation over neighbors (reuse bf16 copy in sA) ----
    {
        int h = tid & 63, g = tid >> 6;          // 4 groups x 25 neighbors
        float s = 0.f;
        for (int k = g * 25; k < g * 25 + 25; ++k)
            s += sWgt[k] * bf2f(sA[k * 128 + 64 + h]);
        sAggP[g * 64 + h] = s;
        __syncthreads();
        if (tid < H_)
            aggBuf[(size_t)bl * H_ + tid] =
                sAggP[tid] + sAggP[64 + tid] + sAggP[128 + tid] + sAggP[192 + tid];
    }
}

// ---------------- final kernel: mean over L, head proj, MLP, sigmoid ----------------
__global__ __launch_bounds__(64)
void final_kernel(const float* __restrict__ aggBuf,
                  const int* __restrict__ user_ids,
                  const float* __restrict__ user_table,
                  const float* __restrict__ Wh,   // [4][64][16]
                  const float* __restrict__ bh,   // [64]
                  const float* __restrict__ Wf1,  // [128][64]
                  const float* __restrict__ bf1,  // [64]
                  const float* __restrict__ Wo,   // [64]
                  const float* __restrict__ bo,   // [1]
                  float* __restrict__ out) {
    __shared__ float sAgg[H_], sComb[2 * H_], sF1[H_];
    int b = blockIdx.x, j = threadIdx.x;

    const float* ab = aggBuf + (size_t)b * L_ * H_;
    float s = 0.f;
    for (int l = 0; l < L_; ++l) s += ab[l * H_ + j];
    sAgg[j] = s * (1.f / L_);
    __syncthreads();

    int n = j >> 4, d = j & 15;
    const float* whn = Wh + (size_t)n * H_ * 16 + d;
    float p = bh[j];
    for (int h = 0; h < H_; ++h) p += sAgg[h] * whn[h * 16];

    sComb[j]      = user_table[(size_t)user_ids[b] * H_ + j];
    sComb[H_ + j] = p;
    __syncthreads();

    float f = bf1[j];
    for (int i = 0; i < 2 * H_; ++i) f += sComb[i] * Wf1[i * H_ + j];
    sF1[j] = fmaxf(f, 0.f) * Wo[j];
    __syncthreads();
    for (int st = 32; st > 0; st >>= 1) {
        if (j < st) sF1[j] += sF1[j + st];
        __syncthreads();
    }
    if (j == 0) out[b] = 1.f / (1.f + __expf(-(sF1[0] + bo[0])));
}

// ---------------- launcher ----------------
extern "C" void kernel_launch(void* const* d_in, const int* in_sizes, int n_in,
                              void* d_out, int out_size, void* d_ws, size_t ws_size,
                              hipStream_t stream) {
    const int*   user_ids   = (const int*)  d_in[0];
    const int*   history    = (const int*)  d_in[2];
    const float* neighbor   = (const float*)d_in[3];
    const float* user_table = (const float*)d_in[4];
    const float* item_table = (const float*)d_in[5];
    const float* Wa  = (const float*)d_in[6];
    const float* ba  = (const float*)d_in[7];
    const float* W1  = (const float*)d_in[8];
    const float* b1  = (const float*)d_in[9];
    const float* W2  = (const float*)d_in[10];
    const float* b2  = (const float*)d_in[11];
    const float* W3  = (const float*)d_in[12];
    const float* b3  = (const float*)d_in[13];
    const float* Wh  = (const float*)d_in[14];
    const float* bh  = (const float*)d_in[15];
    const float* Wf1 = (const float*)d_in[16];
    const float* bf1 = (const float*)d_in[17];
    const float* Wo  = (const float*)d_in[18];
    const float* bo  = (const float*)d_in[19];
    float* out = (float*)d_out;

    // workspace layout: packed weights contiguous, then agg buffer
    unsigned short* pWa = (unsigned short*)d_ws;
    unsigned short* pW1 = (unsigned short*)((char*)d_ws + PWA_B);
    unsigned short* pW2 = (unsigned short*)((char*)d_ws + PWA_B + PW1_B);
    float*       aggBuf = (float*)((char*)d_ws + PW_TOTAL);             // 12800*64 f32

    { int tot = 4 * 6 * 512; pack_b_kernel<<<(tot + 255) / 256, 256, 0, stream>>>(Wa, pWa, 128, 96, 6, tot); }
    { int tot = 3 * 3 * 512; pack_b_kernel<<<(tot + 255) / 256, 256, 0, stream>>>(W1, pW1, 96, 48, 3, tot); }
    { int tot = 2 * 1 * 512; pack_b_kernel<<<(tot + 255) / 256, 256, 0, stream>>>(W2, pW2, 48, 16, 1, tot); }

    edge_mlp_kernel<<<B_ * L_, 256, SMEM_TOTAL, stream>>>(history, item_table, neighbor,
                                                          (const unsigned int*)d_ws,
                                                          ba, b1, b2, W3, b3, aggBuf);

    final_kernel<<<B_, 64, 0, stream>>>(aggBuf, user_ids, user_table,
                                        Wh, bh, Wf1, bf1, Wo, bo, out);
}
